// CrissCrossAttention_77695958385414
// MI455X (gfx1250) — matrix operand order repaired and checked by reading, not compiled, not run on hardware
//
#include <hip/hip_runtime.h>

// Criss-Cross Attention on MI455X (gfx1250): wave32, bf16 WMMA (f32 accum),
// TDM (tensor_load_to_lds) double-buffered LDS staging for the big GEMM,
// transposed v copy so both aggregation GEMMs gather contiguous bf16 pairs.
// B=8, C=512, H=W=64, Co=64.

#define HW 4096
#define CIN 512

typedef __attribute__((ext_vector_type(16))) __bf16 v16bf;
typedef __attribute__((ext_vector_type(8)))  int    v8i;
typedef __attribute__((ext_vector_type(8)))  float  v8f;
typedef __attribute__((ext_vector_type(4)))  unsigned su4;
typedef __attribute__((ext_vector_type(8)))  unsigned su8;

union Frag { v8i i; v16bf b; };

__device__ __forceinline__ unsigned short f2bf(float f) {
  unsigned u = __float_as_uint(f);
  return (unsigned short)((u + 0x7FFFu + ((u >> 16) & 1u)) >> 16);
}

// K offsets per VGPR for the 16-bit WMMA operand layouts (wave32).
// A 16x32: lane m=L&15; v0..3 -> K={0..7}+8*(L>=16), v4..7 -> +16.
// B 32x16: lane n=L&15; same K mapping as A.
__device__ __forceinline__ void frag_offsets(int lane, int* ka, int* kb) {
  int hiA = (lane >> 4) << 3;
  int hiB = (lane >> 4) << 4;
#pragma unroll
  for (int v = 0; v < 8; ++v) {
    ka[v] = ((v & 3) * 2) + hiA + ((v >> 2) << 4);
    kb[v] = ((v & 3) * 2) + hiA + ((v >> 2) << 4);
  }
}

__device__ __forceinline__ v8f wmma_bf16(const Frag& a, const Frag& b, v8f c) {
  return __builtin_amdgcn_wmma_f32_16x16x32_bf16(false, a.b, false, b.b,
                                                 (short)0, c, false, false);
}

// ---------------------------------------------------------------------------
// TDM: 2D bf16 tile (tile0 x tile1) global -> LDS, with 1-DWORD pad per
// 16-DWORD (32 bf16) row => LDS row stride 34 bf16 (conflict-free gathers).
// D# per CDNA5 ISA 8.3/8.4; groups 2/3 disabled (2D tensor).
// ---------------------------------------------------------------------------
__device__ __forceinline__ void tdm_load_tile_bf16(
    unsigned lds_byte_addr, unsigned long long gaddr_bytes,
    unsigned td0, unsigned td1, unsigned tile0, unsigned tile1,
    unsigned stride0) {
  su4 g0;
  g0[0] = 1u;                                           // count=1, user mode
  g0[1] = lds_byte_addr;                                // lds_addr
  g0[2] = (unsigned)gaddr_bytes;                        // global_addr[31:0]
  g0[3] = ((unsigned)(gaddr_bytes >> 32) & 0x01FFFFFFu) // global_addr[56:32]
          | 0x80000000u;                                // type=2 ("image")
  su8 g1;
  g1[0] = (1u << 16)            // data_size: 2 bytes
        | (1u << 20)            // pad_enable
        | (3u << 22);           // pad_interval: 16 DWORDs (pad_amount=0 -> 1 DWORD)
  g1[1] = (td0 & 0xFFFFu) << 16;                        // tensor_dim0[15:0]
  g1[2] = (td0 >> 16) | ((td1 & 0xFFFFu) << 16);        // td0 hi | td1 lo
  g1[3] = (td1 >> 16) | (tile0 << 16);                  // td1 hi | tile_dim0
  g1[4] = tile1;                                        // tile_dim1 (tile_dim2=0)
  g1[5] = stride0;                                      // tensor_dim0_stride lo
  g1[6] = 0u;                                           // stride hi | dim1_stride lo
  g1[7] = 0u;
  asm volatile("tensor_load_to_lds %0, %1" :: "s"(g0), "s"(g1) : "memory");
}

// ---------------------------------------------------------------------------
// Pre-pass 1: pack Wall bf16 [640][512] = [Wq(64); Wk(64); Wv(512)].
// ---------------------------------------------------------------------------
__global__ __launch_bounds__(256) void packw_kernel(
    const float* __restrict__ Wq, const float* __restrict__ Wk,
    const float* __restrict__ Wv, unsigned short* __restrict__ wall) {
  int idx = blockIdx.x * 256 + threadIdx.x;  // 640*512 = 327680 = 1280*256
  int row = idx >> 9, col = idx & 511;
  const float* src = (row < 64)  ? (Wq + row * CIN)
                   : (row < 128) ? (Wk + (row - 64) * CIN)
                                 : (Wv + (row - 128) * CIN);
  wall[(size_t)row * CIN + col] = f2bf(src[col]);
}

// ---------------------------------------------------------------------------
// Pre-pass 2: x (b,c,n) f32 -> xT (b,n,c) bf16 via 64x64 LDS tile transpose.
// ---------------------------------------------------------------------------
__global__ __launch_bounds__(256) void xt_kernel(
    const float* __restrict__ x, unsigned short* __restrict__ xT) {
  __shared__ __align__(16) unsigned short tile[64][65];
  const int n0 = blockIdx.x * 64, c0 = blockIdx.y * 64, b = blockIdx.z;
  const int t = threadIdx.x;
  const float* xb = x + (size_t)b * CIN * HW;
#pragma unroll
  for (int r = 0; r < 16; ++r) {
    int idx = r * 256 + t;
    int c = idx >> 6, n = idx & 63;
    tile[c][n] = f2bf(xb[(size_t)(c0 + c) * HW + n0 + n]);
  }
  __syncthreads();
  unsigned short* xtb = xT + (size_t)b * HW * CIN;
#pragma unroll
  for (int r = 0; r < 16; ++r) {
    int idx = r * 256 + t;
    int n = idx >> 6, c = idx & 63;
    xtb[(size_t)(n0 + n) * CIN + c0 + c] = tile[c][n];
  }
}

// ---------------------------------------------------------------------------
// Stage 1: fused QKV projection GEMM, TDM-staged, double-buffered.
// Out[o,n] = sum_c Wall[o,c]*xT[n,c] + bias.  Tile 128(M) x 64(N), K-step 32.
// ---------------------------------------------------------------------------
__global__ __launch_bounds__(256) void qkv_kernel(
    const unsigned short* __restrict__ wall,  // [640][512] bf16
    const unsigned short* __restrict__ xT,    // [b][4096][512] bf16
    const float* __restrict__ bq, const float* __restrict__ bk,
    const float* __restrict__ bv,
    unsigned short* __restrict__ qo, unsigned short* __restrict__ ko,
    unsigned short* __restrict__ vo) {
  const int nb = blockIdx.x;   // 64 N-blocks
  const int mb = blockIdx.y;   // 5 M-blocks of 128
  const int b  = blockIdx.z;   // batch
  const int t = threadIdx.x, lane = t & 31, wv = t >> 5;
  const int wm = wv >> 1, wn = wv & 1;  // 4x2 wave grid: 32 rows x 32 cols

  __shared__ __align__(16) unsigned short As[2][128 * 34];  // TDM-padded
  __shared__ __align__(16) unsigned short Bs[2][64 * 34];   // xT rows [n][k]

  const int m0 = mb * 128, n0 = nb * 64;

  int ka[8], kb[8];
  frag_offsets(lane, ka, kb);
  const int mrow = lane & 15;

  const v8f vzero = {0.f, 0.f, 0.f, 0.f, 0.f, 0.f, 0.f, 0.f};
  v8f acc[2][2];
#pragma unroll
  for (int i = 0; i < 2; ++i) { acc[i][0] = vzero; acc[i][1] = vzero; }

  auto issue = [&](int buf, int k0) {
    unsigned long long ga = (unsigned long long)(size_t)wall +
                            ((size_t)m0 * CIN + k0) * 2ull;
    tdm_load_tile_bf16((unsigned)(size_t)&As[buf][0], ga,
                       CIN, 640, 32, 128, CIN);
    unsigned long long gb = (unsigned long long)(size_t)xT +
                            (((size_t)b * HW + n0) * CIN + k0) * 2ull;
    tdm_load_tile_bf16((unsigned)(size_t)&Bs[buf][0], gb,
                       CIN, HW, 32, 64, CIN);
  };

  if (wv == 0) issue(0, 0);          // prologue: fill buffer 0

  for (int ks = 0; ks < CIN / 32; ++ks) {
    const int cur = ks & 1;
    if (wv == 0) {
      __builtin_amdgcn_s_wait_tensorcnt(0);     // buffer `cur` ready
      if (ks + 1 < CIN / 32) issue(cur ^ 1, 32 * (ks + 1));  // prefetch next
    }
    __syncthreads();

    Frag a[2], bfr[2];
#pragma unroll
    for (int mt = 0; mt < 2; ++mt) {
      int mm = wm * 32 + mt * 16 + mrow;
#pragma unroll
      for (int v = 0; v < 8; ++v)
        a[mt].i[v] = *(const int*)&As[cur][mm * 34 + ka[v]];
    }
#pragma unroll
    for (int nt = 0; nt < 2; ++nt) {
      int nn = wn * 32 + nt * 16 + mrow;
#pragma unroll
      for (int v = 0; v < 8; ++v)
        bfr[nt].i[v] = *(const int*)&Bs[cur][nn * 34 + kb[v]];
    }
#pragma unroll
    for (int mt = 0; mt < 2; ++mt)
#pragma unroll
      for (int nt = 0; nt < 2; ++nt)
        acc[mt][nt] = wmma_bf16(a[mt], bfr[nt], acc[mt][nt]);
    __syncthreads();   // all waves done reading `cur` before it is refilled
  }

  // Epilogue: segment select is uniform per 16-row tile (boundaries 16-aligned)
#pragma unroll
  for (int mt = 0; mt < 2; ++mt) {
    const int tbase = m0 + wm * 32 + mt * 16;   // wave-uniform
    unsigned short* dst;
    const float* bias;
    int rbase, sbase;
    if (tbase < 64)       { dst = qo; bias = bq; rbase = tbase;       sbase = b * 64; }
    else if (tbase < 128) { dst = ko; bias = bk; rbase = tbase - 64;  sbase = b * 64; }
    else                  { dst = vo; bias = bv; rbase = tbase - 128; sbase = b * 512; }
#pragma unroll
    for (int nt = 0; nt < 2; ++nt) {
      int nn = n0 + wn * 32 + nt * 16 + mrow;
#pragma unroll
      for (int r = 0; r < 8; ++r) {
        int row = rbase + r + ((lane >> 4) << 3);
        dst[(size_t)(sbase + row) * HW + nn] = f2bf(acc[mt][nt][r] + bias[row]);
      }
    }
  }
}

// ---------------------------------------------------------------------------
// Pre-pass 3 (after qkv): vT[b][c][w][h] = v[b][c][h][w]  (64x64 transpose
// per (b,c)) so aggH reads contiguous bf16 pairs along its K dim.
// ---------------------------------------------------------------------------
__global__ __launch_bounds__(256) void vt_kernel(
    const unsigned short* __restrict__ vg, unsigned short* __restrict__ vT) {
  __shared__ __align__(16) unsigned short tile[64][65];
  const int c = blockIdx.x & 511, b = blockIdx.x >> 9;
  const int t = threadIdx.x;
  const unsigned short* src = vg + (size_t)(b * 512 + c) * HW;
  unsigned short* dst = vT + (size_t)(b * 512 + c) * HW;
#pragma unroll
  for (int r = 0; r < 16; ++r) {
    int idx = r * 256 + t;
    int h = idx >> 6, w = idx & 63;
    tile[h][w] = src[h * 64 + w];
  }
  __syncthreads();
#pragma unroll
  for (int r = 0; r < 16; ++r) {
    int idx = r * 256 + t;
    int w = idx >> 6, h = idx & 63;
    dst[w * 64 + h] = tile[h][w];
  }
}

// ---------------------------------------------------------------------------
// Stage 2: energies. mode 0: eH[b,w][h,i] = sum_o q[b,o,h,w] k[b,o,i,w]
//                    mode 1: eW[b,h][w,j] = sum_o q[b,o,h,w] k[b,o,h,j]
// ---------------------------------------------------------------------------
__global__ __launch_bounds__(256) void energy_kernel(
    const unsigned short* __restrict__ qg, const unsigned short* __restrict__ kg,
    float* __restrict__ eH, float* __restrict__ eW) {
  const int mode = blockIdx.y;
  const int b = blockIdx.x >> 6, s = blockIdx.x & 63;
  const int t = threadIdx.x, lane = t & 31, wv = t >> 5;

  __shared__ __align__(16) unsigned short Qs[64 * 66];  // [m][o], padded 66
  __shared__ __align__(16) unsigned short Ks[64 * 66];  // [n][o]

#pragma unroll
  for (int r = 0; r < 16; ++r) {
    int idx = r * 256 + t;
    int mn = idx >> 6, o = idx & 63;
    size_t g = (size_t)(b * 64 + o) * HW + (mode ? (s * 64 + mn) : (mn * 64 + s));
    Qs[mn * 66 + o] = qg[g];
    Ks[mn * 66 + o] = kg[g];
  }
  __syncthreads();

  int ka[8], kb[8];
  frag_offsets(lane, ka, kb);
  const int mrow = lane & 15;
  const int tm = wv >> 1, tnb = (wv & 1) * 32;

  const v8f vzero = {0.f, 0.f, 0.f, 0.f, 0.f, 0.f, 0.f, 0.f};
  v8f acc[2] = {vzero, vzero};
#pragma unroll
  for (int k0 = 0; k0 < 64; k0 += 32) {
    Frag a;
    int mm = tm * 16 + mrow;
#pragma unroll
    for (int v = 0; v < 8; ++v) a.i[v] = *(const int*)&Qs[mm * 66 + k0 + ka[v]];
#pragma unroll
    for (int nt = 0; nt < 2; ++nt) {
      Frag bfr;
      int nn = tnb + nt * 16 + mrow;
#pragma unroll
      for (int v = 0; v < 8; ++v)
        bfr.i[v] = *(const int*)&Ks[nn * 66 + k0 + kb[v]];
      acc[nt] = wmma_bf16(a, bfr, acc[nt]);
    }
  }

  float* eout = mode ? eW : eH;
#pragma unroll
  for (int nt = 0; nt < 2; ++nt) {
    int nn = tnb + nt * 16 + mrow;
#pragma unroll
    for (int r = 0; r < 8; ++r) {
      int mm = tm * 16 + r + ((lane >> 4) << 3);
      eout[((size_t)(b * 64 + s) * 64 + mm) * 64 + nn] = acc[nt][r];
    }
  }
}

// ---------------------------------------------------------------------------
// Stage 3: masked softmax over 128 logits per (b,h,w); one wave per position.
// ---------------------------------------------------------------------------
__global__ __launch_bounds__(256) void softmax_kernel(
    const float* __restrict__ eH, const float* __restrict__ eW,
    unsigned short* __restrict__ aH, unsigned short* __restrict__ aW) {
  const int gw = blockIdx.x * 8 + (threadIdx.x >> 5);  // (b*64+h)*64+w
  const int lane = threadIdx.x & 31;
  const int b = gw >> 12, h = (gw >> 6) & 63, w = gw & 63;
  const float* ehrow = eH + ((size_t)(b * 64 + w) * 64 + h) * 64;
  const float* ewrow = eW + ((size_t)(b * 64 + h) * 64 + w) * 64;
  const int i0 = 2 * lane;
  float vH0 = ehrow[i0], vH1 = ehrow[i0 + 1];
  float vW0 = ewrow[i0], vW1 = ewrow[i0 + 1];
  if (i0 == h)     vH0 = -__builtin_inff();   // diagonal mask on H branch
  if (i0 + 1 == h) vH1 = -__builtin_inff();
  float mx = fmaxf(fmaxf(vH0, vH1), fmaxf(vW0, vW1));
#pragma unroll
  for (int off = 16; off > 0; off >>= 1) mx = fmaxf(mx, __shfl_xor(mx, off, 32));
  float e0 = __expf(vH0 - mx), e1 = __expf(vH1 - mx);
  float e2 = __expf(vW0 - mx), e3 = __expf(vW1 - mx);
  float sm = e0 + e1 + e2 + e3;
#pragma unroll
  for (int off = 16; off > 0; off >>= 1) sm += __shfl_xor(sm, off, 32);
  float inv = 1.0f / sm;
  unsigned int ph = (unsigned int)f2bf(e0 * inv) | ((unsigned int)f2bf(e1 * inv) << 16);
  unsigned int pw = (unsigned int)f2bf(e2 * inv) | ((unsigned int)f2bf(e3 * inv) << 16);
  *(unsigned int*)(aH + (size_t)gw * 64 + i0) = ph;
  *(unsigned int*)(aW + (size_t)gw * 64 + i0) = pw;
}

// ---------------------------------------------------------------------------
// Stage 4a: outH[b,c,h,w] = sum_i v[b,c,i,w]*aH[b,h,w,i]; out = x + g*outH
// A operand from vT[b][c][w][i] -> contiguous aligned bf16 pairs.
// ---------------------------------------------------------------------------
__global__ __launch_bounds__(256) void aggH_kernel(
    const unsigned short* __restrict__ vT, const unsigned short* __restrict__ aH,
    const float* __restrict__ x, const float* __restrict__ gamma,
    float* __restrict__ out) {
  const int b = blockIdx.x >> 6, w = blockIdx.x & 63;
  const int cb = blockIdx.y;
  const int t = threadIdx.x, lane = t & 31, wv = t >> 5;
  const int wm = wv >> 1, wn = wv & 1;
  const int c0 = cb * 128 + wm * 32;
  const int mrow = lane & 15;

  int ka[8], kb[8];
  frag_offsets(lane, ka, kb);

  const v8f vzero = {0.f, 0.f, 0.f, 0.f, 0.f, 0.f, 0.f, 0.f};
  v8f acc[2][2];
#pragma unroll
  for (int i = 0; i < 2; ++i) { acc[i][0] = vzero; acc[i][1] = vzero; }

#pragma unroll
  for (int k0 = 0; k0 < 64; k0 += 32) {
    Frag a[2], bfr[2];
#pragma unroll
    for (int mt = 0; mt < 2; ++mt) {
      const unsigned short* ap =
          vT + (size_t)(b * 512 + c0 + mt * 16 + mrow) * HW + w * 64;
#pragma unroll
      for (int v = 0; v < 8; ++v) a[mt].i[v] = *(const int*)&ap[k0 + ka[v]];
    }
#pragma unroll
    for (int nt = 0; nt < 2; ++nt) {
      int hh = wn * 32 + nt * 16 + mrow;
      const unsigned short* bp = aH + ((size_t)(b * 64 + hh) * 64 + w) * 64;
#pragma unroll
      for (int v = 0; v < 8; ++v) bfr[nt].i[v] = *(const int*)&bp[k0 + kb[v]];
    }
#pragma unroll
    for (int mt = 0; mt < 2; ++mt)
#pragma unroll
      for (int nt = 0; nt < 2; ++nt)
        acc[mt][nt] = wmma_bf16(a[mt], bfr[nt], acc[mt][nt]);
  }

  const float g = gamma[0];
#pragma unroll
  for (int mt = 0; mt < 2; ++mt) {
#pragma unroll
    for (int nt = 0; nt < 2; ++nt) {
      int hh = wn * 32 + nt * 16 + mrow;
#pragma unroll
      for (int r = 0; r < 8; ++r) {
        int c = c0 + mt * 16 + r + ((lane >> 4) << 3);
        size_t o = (size_t)(b * 512 + c) * HW + hh * 64 + w;
        out[o] = x[o] + g * acc[mt][nt][r];
      }
    }
  }
}

// ---------------------------------------------------------------------------
// Stage 4b: out += g * sum_j v[b,c,h,j]*aW[b,h,w,j]
// ---------------------------------------------------------------------------
__global__ __launch_bounds__(256) void aggW_kernel(
    const unsigned short* __restrict__ vg, const unsigned short* __restrict__ aW,
    const float* __restrict__ gamma, float* __restrict__ out) {
  const int b = blockIdx.x >> 6, h = blockIdx.x & 63;
  const int cb = blockIdx.y;
  const int t = threadIdx.x, lane = t & 31, wv = t >> 5;
  const int wm = wv >> 1, wn = wv & 1;
  const int c0 = cb * 128 + wm * 32;
  const int mrow = lane & 15;

  int ka[8], kb[8];
  frag_offsets(lane, ka, kb);

  const v8f vzero = {0.f, 0.f, 0.f, 0.f, 0.f, 0.f, 0.f, 0.f};
  v8f acc[2][2];
#pragma unroll
  for (int i = 0; i < 2; ++i) { acc[i][0] = vzero; acc[i][1] = vzero; }

#pragma unroll
  for (int k0 = 0; k0 < 64; k0 += 32) {
    Frag a[2], bfr[2];
#pragma unroll
    for (int mt = 0; mt < 2; ++mt) {
      const unsigned short* ap =
          vg + (size_t)(b * 512 + c0 + mt * 16 + mrow) * HW + h * 64;
#pragma unroll
      for (int v = 0; v < 8; ++v) a[mt].i[v] = *(const int*)&ap[k0 + ka[v]];
    }
#pragma unroll
    for (int nt = 0; nt < 2; ++nt) {
      int ww = wn * 32 + nt * 16 + mrow;
      const unsigned short* bp = aW + ((size_t)(b * 64 + h) * 64 + ww) * 64;
#pragma unroll
      for (int v = 0; v < 8; ++v) bfr[nt].i[v] = *(const int*)&bp[k0 + kb[v]];
    }
#pragma unroll
    for (int mt = 0; mt < 2; ++mt)
#pragma unroll
      for (int nt = 0; nt < 2; ++nt)
        acc[mt][nt] = wmma_bf16(a[mt], bfr[nt], acc[mt][nt]);
  }

  const float g = gamma[0];
#pragma unroll
  for (int mt = 0; mt < 2; ++mt) {
#pragma unroll
    for (int nt = 0; nt < 2; ++nt) {
      int ww = wn * 32 + nt * 16 + mrow;
#pragma unroll
      for (int r = 0; r < 8; ++r) {
        int c = c0 + mt * 16 + r + ((lane >> 4) << 3);
        size_t o = (size_t)(b * 512 + c) * HW + h * 64 + ww;
        out[o] += g * acc[mt][nt][r];
      }
    }
  }
}

// ---------------------------------------------------------------------------
extern "C" void kernel_launch(void* const* d_in, const int* in_sizes, int n_in,
                              void* d_out, int out_size, void* d_ws, size_t ws_size,
                              hipStream_t stream) {
  (void)in_sizes; (void)n_in; (void)out_size; (void)ws_size;
  const float* x  = (const float*)d_in[0];
  const float* Wq = (const float*)d_in[1];
  const float* bq = (const float*)d_in[2];
  const float* Wk = (const float*)d_in[3];
  const float* bk = (const float*)d_in[4];
  const float* Wv = (const float*)d_in[5];
  const float* bv = (const float*)d_in[6];
  const float* gm = (const float*)d_in[7];
  float* out = (float*)d_out;

  char* ws = (char*)d_ws;
  // Workspace layout (97 MiB total; xT region is reused for vT after qkv):
  unsigned short* qb   = (unsigned short*)(ws);                 //  4 MiB bf16 q
  unsigned short* kb   = (unsigned short*)(ws + (4ull << 20));  //  4 MiB bf16 k
  unsigned short* vb   = (unsigned short*)(ws + (8ull << 20));  // 32 MiB bf16 v
  float*          eH   = (float*)(ws + (40ull << 20));          //  8 MiB f32
  float*          eW   = (float*)(ws + (48ull << 20));          //  8 MiB f32
  unsigned short* aH   = (unsigned short*)(ws + (56ull << 20)); //  4 MiB bf16
  unsigned short* aW   = (unsigned short*)(ws + (60ull << 20)); //  4 MiB bf16
  unsigned short* xT   = (unsigned short*)(ws + (64ull << 20)); // 32 MiB bf16
  unsigned short* vT   = (unsigned short*)(ws + (64ull << 20)); // reuse xT slot
  unsigned short* wall = (unsigned short*)(ws + (96ull << 20)); //  1 MiB bf16

  packw_kernel<<<dim3(1280), 256, 0, stream>>>(Wq, Wk, Wv, wall);
  xt_kernel<<<dim3(64, 8, 8), 256, 0, stream>>>(x, xT);
  qkv_kernel<<<dim3(64, 5, 8), 256, 0, stream>>>(wall, xT, bq, bk, bv,
                                                 qb, kb, vb);
  vt_kernel<<<dim3(4096), 256, 0, stream>>>(vb, vT);   // xT dead from here on
  energy_kernel<<<dim3(512, 2), 256, 0, stream>>>(qb, kb, eH, eW);
  softmax_kernel<<<dim3(4096), 256, 0, stream>>>(eH, eW, aH, aW);
  aggH_kernel<<<dim3(512, 4), 256, 0, stream>>>(vT, aH, x, gm, out);
  aggW_kernel<<<dim3(512, 4), 256, 0, stream>>>(vb, aW, gm, out);
}
